// Net_76622216561354
// MI455X (gfx1250) — matrix operand order repaired and checked by reading, not compiled, not run on hardware
//
#include <hip/hip_runtime.h>
#include <hip/hip_bf16.h>
#include <stdint.h>

// ---------------------------------------------------------------------------
// MPNN (NNConv + GRU) for MI455X / gfx1250, wave32, bf16 WMMA.
//
//   msg_e = xi[src] @ (ea_e@nn1_w + nn1_b)  ==  sum_k ea[e,k]*t_k[src] + t_b[src]
//   with t_g = out @ (lin1_w@Wg) + (lin1_b@Wg)   (Wg: 5 shared 16x16 mats)
//   -> GEMM1: [N,64] x [64,96] bf16 WMMA  (t0..t3, t_bias, root-term)
//   GRU: one GEMM2 [N, 96(pad of 80)] x [96,256] bf16 WMMA producing
//   (r | z | i_n | h_n), fused sigmoid/tanh epilogue writes h in place.
// B fragments are staged into LDS once per workgroup via the gfx1250
// async global->LDS path (ASYNCcnt) when available. Working set/iter < L2.
// ---------------------------------------------------------------------------

typedef __attribute__((ext_vector_type(16))) __bf16 v16bf;
typedef __attribute__((ext_vector_type(8)))  float  v8f;
typedef int v4i_ __attribute__((vector_size(16)));

constexpr int NN   = 131072;   // nodes
constexpr int NE   = 524288;   // edges
constexpr int NGR  = 8192;     // graphs

// workspace offsets (bytes)
constexpr size_t O_H     = 0;                         // [N,64] f32
constexpr size_t O_T     = O_H     + (size_t)NN*64*4; // [N,96] f32
constexpr size_t O_AGGR  = O_T     + (size_t)NN*96*4; // [N,16] f32
constexpr size_t O_INV   = O_AGGR  + (size_t)NN*16*4; // [N]    f32
constexpr size_t O_CNT   = O_INV   + (size_t)NN*4;    // [N]    f32
constexpr size_t O_B1F   = O_CNT   + (size_t)NN*4;    // 2*6*32*8 u32
constexpr size_t O_BIAS1 = O_B1F   + 12288;           // 96 f32 (pad 512)
constexpr size_t O_B2F   = O_BIAS1 + 512;             // 4*3*16*32*8 u32
constexpr size_t O_BIAS2 = O_B2F   + 196608;          // 4*256 f32

// --- optional CDNA5 async global->LDS copy path ----------------------------
#if defined(__has_builtin)
#  if __has_builtin(__builtin_amdgcn_global_load_async_to_lds_b128)
#    define ASYNC_LDS 1
#  endif
#endif
#ifndef ASYNC_LDS
#  define ASYNC_LDS 0
#endif

__device__ inline void wait_asynccnt0() {
#if defined(__has_builtin)
#  if __has_builtin(__builtin_amdgcn_s_wait_asynccnt)
  __builtin_amdgcn_s_wait_asynccnt(0);
#  else
  asm volatile("s_wait_asynccnt 0x0" ::: "memory");
#  endif
#endif
}

// Cooperative copy of NU4 uint4 per thread from global to LDS (blockDim == 128).
template <int NU4>
__device__ inline void stage_lds(uint32_t* __restrict__ sdst,
                                 const uint32_t* __restrict__ gsrc) {
  const uint4* gs = (const uint4*)gsrc;
  uint4* ls = (uint4*)sdst;
#if ASYNC_LDS
#pragma unroll
  for (int i = 0; i < NU4; ++i) {
    int idx = (int)threadIdx.x + i * 128;
    __builtin_amdgcn_global_load_async_to_lds_b128(
        (__attribute__((address_space(1))) v4i_*)(v4i_*)(gs + idx),
        (__attribute__((address_space(3))) v4i_*)(v4i_*)(ls + idx), 0, 0);
  }
  wait_asynccnt0();
#else
#pragma unroll
  for (int i = 0; i < NU4; ++i) {
    int idx = (int)threadIdx.x + i * 128;
    ls[idx] = gs[idx];
  }
#endif
  __syncthreads();
}

__device__ inline uint32_t pack_bf16(float a, float b) {
  __bf16 x = (__bf16)a, y = (__bf16)b;
  uint16_t ux = __builtin_bit_cast(uint16_t, x);
  uint16_t uy = __builtin_bit_cast(uint16_t, y);
  return (uint32_t)ux | ((uint32_t)uy << 16);
}
__device__ inline float sig_(float x) { return 1.0f / (1.0f + __expf(-x)); }
__device__ inline float tanh_(float x) {
  x = fminf(fmaxf(x, -15.0f), 15.0f);
  float e = __expf(2.0f * x);
  return (e - 1.0f) / (e + 1.0f);
}

// ------------------------- precompute kernels ------------------------------

// B1frag: [kstep(2)][tile(6)][lane(32)][vgpr(8)] packed bf16 pairs for
// B = (lin1_w @ Wg) stacked over 6 groups of 16 cols. bias1[96] appended.
// B-frag element (lane,v,s): k = ks*32 + (lane>>4)*8 + (v&3)*2 + (v>>2)*16 + s, n = tile*16+(lane&15)
__global__ void prep1_kernel(const float* __restrict__ lin1_w, const float* __restrict__ lin1_b,
                             const float* __restrict__ nn1_w,  const float* __restrict__ nn1_b,
                             const float* __restrict__ root_w, const float* __restrict__ conv_b,
                             uint32_t* __restrict__ B1frag, float* __restrict__ bias1) {
  int idx = blockIdx.x * blockDim.x + threadIdx.x;
  const int NFRAG = 2 * 6 * 32 * 8;
  if (idx < NFRAG) {
    int v = idx & 7, lane = (idx >> 3) & 31, t = (idx >> 8) % 6, ks = (idx >> 8) / 6;
    float val[2];
    for (int s = 0; s < 2; ++s) {
      int k = ks * 32 + (lane >> 4) * 8 + (v & 3) * 2 + (v >> 2) * 16 + s;
      int c = t * 16 + (lane & 15);
      int g = c >> 4, o = c & 15;
      float acc = 0.0f;
      if (k < 64) {
        for (int i = 0; i < 16; ++i) {
          float w = lin1_w[k * 16 + i];
          float mm = (g < 4) ? nn1_w[g * 256 + i * 16 + o]
                   : (g == 4) ? nn1_b[i * 16 + o]
                              : root_w[i * 16 + o];
          acc += w * mm;
        }
      }
      val[s] = acc;
    }
    B1frag[idx] = pack_bf16(val[0], val[1]);
  } else if (idx < NFRAG + 96) {
    int c = idx - NFRAG;
    int g = c >> 4, o = c & 15;
    float acc = 0.0f;
    for (int i = 0; i < 16; ++i) {
      float mm = (g < 4) ? nn1_w[g * 256 + i * 16 + o]
               : (g == 4) ? nn1_b[i * 16 + o]
                          : root_w[i * 16 + o];
      acc += lin1_b[i] * mm;
    }
    if (g == 5) acc += conv_b[o];
    bias1[c] = acc;
  }
}

// B2frag: [j(4)][kstep(3)][tile(16)][lane(32)][vgpr(8)], combined GRU weight
// rows 0..15 = xin (W_ih^T), rows 16..79 = h (W_hh^T), 80..95 = 0.
// cols: [0..63]=r (ih+hh), [64..127]=z (ih+hh), [128..191]=i_n (ih), [192..255]=h_n (hh)
__global__ void prep2_kernel(const float* __restrict__ w_ih, const float* __restrict__ w_hh,
                             const float* __restrict__ b_ih, const float* __restrict__ b_hh,
                             uint32_t* __restrict__ B2frag, float* __restrict__ bias2) {
  int idx = blockIdx.x * blockDim.x + threadIdx.x;
  const int NFRAG = 4 * 3 * 16 * 32 * 8;
  if (idx < NFRAG) {
    int v = idx & 7, lane = (idx >> 3) & 31, t = (idx >> 8) & 15;
    int jk = idx >> 12, ks = jk % 3, j = jk / 3;
    float val[2];
    for (int s = 0; s < 2; ++s) {
      int k = ks * 32 + (lane >> 4) * 8 + (v & 3) * 2 + (v >> 2) * 16 + s;
      int c = t * 16 + (lane & 15);
      int grp = c >> 6, d = c & 63;
      float w = 0.0f;
      if (k < 16) {                       // xin rows -> W_ih
        if (grp != 3) {
          int gc = (grp == 2) ? (128 + d) : (grp * 64 + d);
          w = w_ih[(size_t)(j * 192 + gc) * 16 + k];
        }
      } else if (k < 80) {                // h rows -> W_hh
        if (grp != 2) {
          int gc = (grp == 3) ? (128 + d) : (grp * 64 + d);
          w = w_hh[(size_t)(j * 192 + gc) * 64 + (k - 16)];
        }
      }
      val[s] = w;
    }
    B2frag[idx] = pack_bf16(val[0], val[1]);
  } else if (idx < NFRAG + 4 * 256) {
    int r = idx - NFRAG;
    int j = r >> 8, c = r & 255;
    int grp = c >> 6, d = c & 63;
    float b;
    if (grp == 0)      b = b_ih[j * 192 + d]       + b_hh[j * 192 + d];
    else if (grp == 1) b = b_ih[j * 192 + 64 + d]  + b_hh[j * 192 + 64 + d];
    else if (grp == 2) b = b_ih[j * 192 + 128 + d];
    else               b = b_hh[j * 192 + 128 + d];
    bias2[r] = b;
  }
}

__global__ void degree_kernel(const int* __restrict__ dst, float* __restrict__ cnt) {
  int e = blockIdx.x * blockDim.x + threadIdx.x;
  if (e < NE) atomicAdd(&cnt[dst[e]], 1.0f);
}
__global__ void inv_kernel(const float* __restrict__ cnt, float* __restrict__ inv) {
  int n = blockIdx.x * blockDim.x + threadIdx.x;
  if (n < NN) inv[n] = 1.0f / fmaxf(cnt[n], 1.0f);
}

// lin0 + relu: h = relu(x @ lin0_w + lin0_b)   [N,16]x[16,64]
__global__ void lin0_kernel(const float* __restrict__ x, const float* __restrict__ w,
                            const float* __restrict__ b, float* __restrict__ h) {
  int tid = blockIdx.x * blockDim.x + threadIdx.x;
  int n = tid >> 6, d = tid & 63;
  float acc = b[d];
#pragma unroll
  for (int i = 0; i < 16; ++i) acc += x[(size_t)n * 16 + i] * w[i * 64 + d];
  h[(size_t)n * 64 + d] = fmaxf(acc, 0.0f);
}

// ------------------------------ GEMM kernels -------------------------------

// ISA 16-bit A 16x32 layout: lane = half*16 + m; VGPR v holds
// K = kstep*32 + half*8 + (v&3)*2 + (v>>2)*16  (pair K,K+1 packed).
__device__ inline v16bf buildA_from_h(const float* __restrict__ hrow, int kstep, int half) {
  v16bf a;
#pragma unroll
  for (int v = 0; v < 8; ++v) {
    int kb = kstep * 32 + half * 8 + (v & 3) * 2 + (v >> 2) * 16;
    float2 p = *(const float2*)(hrow + kb);
    a[2 * v]     = (__bf16)p.x;
    a[2 * v + 1] = (__bf16)p.y;
  }
  return a;
}

// GEMM1: T[N,96] = h[N,64] @ B1 + bias1.  1 wave = 16 rows x 96 cols (6 tiles).
__global__ __launch_bounds__(128) void
gemm1_kernel(const float* __restrict__ h, const uint32_t* __restrict__ B1frag,
             const float* __restrict__ bias1, float* __restrict__ T) {
  __shared__ uint32_t sB[2 * 6 * 32 * 8];          // 12 KB
  stage_lds<6>(sB, B1frag);                        // 6 uint4 per thread

  int lane = threadIdx.x & 31, wave = threadIdx.x >> 5;
  int half = lane >> 4, m = lane & 15, n = lane & 15;
  int row0 = blockIdx.x * 64 + wave * 16;
  const float* hrow = h + (size_t)(row0 + m) * 64;

  v16bf a0 = buildA_from_h(hrow, 0, half);
  v16bf a1 = buildA_from_h(hrow, 1, half);

  v8f acc[6] = {};
#pragma unroll
  for (int t = 0; t < 6; ++t) {
    v16bf b0 = *(const v16bf*)(sB + ((0 * 6 + t) * 32 + lane) * 8);
    acc[t] = __builtin_amdgcn_wmma_f32_16x16x32_bf16(false, a0, false, b0, (short)0, acc[t], false, false);
    v16bf b1 = *(const v16bf*)(sB + ((1 * 6 + t) * 32 + lane) * 8);
    acc[t] = __builtin_amdgcn_wmma_f32_16x16x32_bf16(false, a1, false, b1, (short)0, acc[t], false, false);
  }
#pragma unroll
  for (int t = 0; t < 6; ++t) {
    int col = t * 16 + n;
    float bv = bias1[col];
#pragma unroll
    for (int r = 0; r < 8; ++r) {
      int orow = row0 + half * 8 + r;             // D layout: row = half*8 + r
      T[(size_t)orow * 96 + col] = acc[t][r] + bv;
    }
  }
}

// edge message + scatter: msg = t_b[src] + sum_k ea_k * t_k[src]; atomic add to aggr[dst]
__global__ void edge_kernel(const float* __restrict__ T, const float* __restrict__ ea,
                            const int* __restrict__ ei, float* __restrict__ aggr) {
  int tid = blockIdx.x * blockDim.x + threadIdx.x;
  int e = tid >> 4, o = tid & 15;
  int src = ei[e];
  int dst = ei[NE + e];
  const float* Ts = T + (size_t)src * 96;
  float msg = Ts[64 + o];
#pragma unroll
  for (int k = 0; k < 4; ++k) msg += ea[(size_t)e * 4 + k] * Ts[k * 16 + o];
  atomicAdd(&aggr[(size_t)dst * 16 + o], msg);
}

// GEMM2 + fused GRU: A = [xin | h | 0] (K=96), B = [96,256] -> (r|z|i_n|h_n),
// epilogue updates h in place. 1 wave = 16 rows, 16 accumulator tiles.
__global__ __launch_bounds__(128) void
gemm2_gru_kernel(float* __restrict__ h, const float* __restrict__ T,
                 const float* __restrict__ aggr, const float* __restrict__ inv_cnt,
                 const uint32_t* __restrict__ B2frag, const float* __restrict__ bias2) {
  __shared__ uint32_t sB[3 * 16 * 32 * 8];         // 48 KB
  stage_lds<24>(sB, B2frag);                       // 24 uint4 per thread

  int lane = threadIdx.x & 31, wave = threadIdx.x >> 5;
  int half = lane >> 4, m = lane & 15, n = lane & 15;
  int row0 = blockIdx.x * 64 + wave * 16;
  int arow = row0 + m;
  float invc = inv_cnt[arow];
  const float* Trow = T + (size_t)arow * 96;
  const float* Grow = aggr + (size_t)arow * 16;
  const float* Hrow = h + (size_t)arow * 64;

  v16bf a[3];
#pragma unroll
  for (int ks = 0; ks < 3; ++ks) {
#pragma unroll
    for (int v = 0; v < 8; ++v) {
      int kb = ks * 32 + half * 8 + (v & 3) * 2 + (v >> 2) * 16;
      float x0, x1;
      if (kb < 16) {                       // xin = root-term + mean-aggr
        float2 t2 = *(const float2*)(Trow + 80 + kb);
        float2 g2 = *(const float2*)(Grow + kb);
        x0 = t2.x + g2.x * invc;
        x1 = t2.y + g2.y * invc;
      } else if (kb < 80) {                // hidden state
        float2 h2 = *(const float2*)(Hrow + (kb - 16));
        x0 = h2.x; x1 = h2.y;
      } else {                             // zero pad K 80..95
        x0 = 0.0f; x1 = 0.0f;
      }
      a[ks][2 * v]     = (__bf16)x0;
      a[ks][2 * v + 1] = (__bf16)x1;
    }
  }

  v8f acc[16] = {};
#pragma unroll
  for (int t = 0; t < 16; ++t) {
#pragma unroll
    for (int ks = 0; ks < 3; ++ks) {
      v16bf b = *(const v16bf*)(sB + ((ks * 16 + t) * 32 + lane) * 8);
      acc[t] = __builtin_amdgcn_wmma_f32_16x16x32_bf16(false, a[ks], false, b, (short)0, acc[t], false, false);
    }
  }

#pragma unroll
  for (int tt = 0; tt < 4; ++tt) {
    int d = tt * 16 + n;
    float b_r  = bias2[d];
    float b_z  = bias2[64 + d];
    float b_in = bias2[128 + d];
    float b_hn = bias2[192 + d];
#pragma unroll
    for (int r = 0; r < 8; ++r) {
      int orow = row0 + half * 8 + r;
      float gr = sig_(acc[tt][r] + b_r);
      float gz = sig_(acc[4 + tt][r] + b_z);
      float nc = tanh_(acc[8 + tt][r] + b_in + gr * (acc[12 + tt][r] + b_hn));
      float hold = h[(size_t)orow * 64 + d];
      h[(size_t)orow * 64 + d] = (1.0f - gz) * nc + gz * hold;
    }
  }
}

// y = h @ lin2_w ; segment-sum into pooled[NG]
__global__ void pool_kernel(const float* __restrict__ h, const float* __restrict__ w,
                            const int* __restrict__ batch, float* __restrict__ out) {
  int n = blockIdx.x * blockDim.x + threadIdx.x;
  if (n >= NN) return;
  const float4* hv = (const float4*)(h + (size_t)n * 64);
  float acc = 0.0f;
#pragma unroll
  for (int q = 0; q < 16; ++q) {
    float4 hh = hv[q];
    acc += hh.x * w[4 * q] + hh.y * w[4 * q + 1] + hh.z * w[4 * q + 2] + hh.w * w[4 * q + 3];
  }
  atomicAdd(&out[batch[n]], acc);
}

// ------------------------------- launcher ----------------------------------

extern "C" void kernel_launch(void* const* d_in, const int* in_sizes, int n_in,
                              void* d_out, int out_size, void* d_ws, size_t ws_size,
                              hipStream_t stream) {
  const float* x         = (const float*)d_in[0];
  const float* edge_attr = (const float*)d_in[1];
  const float* lin0_w    = (const float*)d_in[2];
  const float* lin0_b    = (const float*)d_in[3];
  const float* nn1_w     = (const float*)d_in[4];
  const float* nn1_b     = (const float*)d_in[5];
  const float* root_w    = (const float*)d_in[6];
  const float* conv_b    = (const float*)d_in[7];
  const float* gru_w_ih  = (const float*)d_in[8];
  const float* gru_w_hh  = (const float*)d_in[9];
  const float* gru_b_ih  = (const float*)d_in[10];
  const float* gru_b_hh  = (const float*)d_in[11];
  const float* lin1_w    = (const float*)d_in[12];
  const float* lin1_b    = (const float*)d_in[13];
  const float* lin2_w    = (const float*)d_in[14];
  const int*   edge_idx  = (const int*)d_in[15];
  const int*   batch     = (const int*)d_in[16];
  float* out = (float*)d_out;

  char* ws = (char*)d_ws;
  float*    h      = (float*)(ws + O_H);
  float*    T      = (float*)(ws + O_T);
  float*    aggr   = (float*)(ws + O_AGGR);
  float*    invc   = (float*)(ws + O_INV);
  float*    cnt    = (float*)(ws + O_CNT);
  uint32_t* B1frag = (uint32_t*)(ws + O_B1F);
  float*    bias1  = (float*)(ws + O_BIAS1);
  uint32_t* B2frag = (uint32_t*)(ws + O_B2F);
  float*    bias2  = (float*)(ws + O_BIAS2);

  // degree -> inverse counts
  (void)hipMemsetAsync(cnt, 0, (size_t)NN * 4, stream);
  degree_kernel<<<NE / 256, 256, 0, stream>>>(edge_idx + NE, cnt);
  inv_kernel<<<NN / 256, 256, 0, stream>>>(cnt, invc);

  // weight prep (bf16 fragment swizzle)
  prep1_kernel<<<13, 256, 0, stream>>>(lin1_w, lin1_b, nn1_w, nn1_b, root_w, conv_b,
                                       B1frag, bias1);
  prep2_kernel<<<197, 256, 0, stream>>>(gru_w_ih, gru_w_hh, gru_b_ih, gru_b_hh,
                                        B2frag, bias2);

  // h = relu(x @ lin0_w + b)
  lin0_kernel<<<(NN * 64) / 256, 256, 0, stream>>>(x, lin0_w, lin0_b, h);

  // 4 GRUs x 2 inner repeats
  for (int it = 0; it < 8; ++it) {
    int j = it >> 1;
    gemm1_kernel<<<NN / 64, 128, 0, stream>>>(h, B1frag, bias1, T);
    (void)hipMemsetAsync(aggr, 0, (size_t)NN * 16 * 4, stream);
    edge_kernel<<<(NE * 16) / 256, 256, 0, stream>>>(T, edge_attr, edge_idx, aggr);
    gemm2_gru_kernel<<<NN / 64, 128, 0, stream>>>(h, T, aggr, invc,
                                                  B2frag + (size_t)j * 3 * 16 * 32 * 8,
                                                  bias2 + (size_t)j * 256);
  }

  // readout + global_add_pool
  (void)hipMemsetAsync(out, 0, (size_t)NGR * 4, stream);
  pool_kernel<<<NN / 256, 256, 0, stream>>>(h, lin2_w, batch, out);
}
